// DeepFM_36498632081512
// MI455X (gfx1250) — matrix operand order repaired and checked by reading, not compiled, not run on hardware
//
#include <hip/hip_runtime.h>

typedef _Float16 half_t;
typedef __attribute__((ext_vector_type(8)))  _Float16 v8h;
typedef __attribute__((ext_vector_type(16))) _Float16 v16h;
typedef __attribute__((ext_vector_type(8)))  float    v8f;

#define ROWS 128          // rows per workgroup
#define NROWS (128*1024)  // B*G
#define FVOC 10000
#define K1 288            // 269 padded to 9*32
#define N1 256
#define K2 256
#define N2 128
#define K3 160            // 130 padded to 5*32
#define N3 128

// LDS layout (bytes)
#define A1_OFF 0
#define A2_OFF (ROWS*K1*2)              // 73728
#define FM_OFF (A2_OFF + ROWS*N1*2)     // 139264
#define LDS_BYTES (FM_OFF + ROWS*2*4)   // 140288

// workspace layout (bytes): packed f16 weights, [N][Kpad]
#define W1P_OFF 0
#define W2P_OFF (W1P_OFF + N1*K1*2)     // 147456
#define W3P_OFF (W2P_OFF + N2*K2*2)     // 212992
#define WS_NEED (W3P_OFF + N3*K3*2)     // 253952

// ---------------------------------------------------------------------------
// Weight packing: fp32 [K][N] -> f16 [N][Kpad], zero-padded K tail
// ---------------------------------------------------------------------------
__global__ __launch_bounds__(256)
void pack_weights(const float* __restrict__ w1, const float* __restrict__ w2,
                  const float* __restrict__ wo,
                  half_t* __restrict__ W1p, half_t* __restrict__ W2p,
                  half_t* __restrict__ W3p) {
  int i = blockIdx.x * 256 + threadIdx.x;
  if (i < N1 * K1) {
    int n = i / K1, k = i % K1;
    W1p[i] = (k < 269) ? (half_t)w1[(size_t)k * N1 + n] : (half_t)0.f;
  }
  if (i < N2 * K2) {
    int n = i / K2, k = i % K2;
    W2p[i] = (half_t)w2[(size_t)k * N2 + n];
  }
  if (i < N3 * K3) {
    int n = i / K3, k = i % K3;
    W3p[i] = (k < 130) ? (half_t)wo[(size_t)k * N3 + n] : (half_t)0.f;
  }
}

// ---------------------------------------------------------------------------
// Fragment loaders (wave32, 16x16x32 f16 WMMA)
// A (16xK, row in lane): lanes 0-15 hold K 0..15, lanes 16-31 hold
// K 16..31 for row M = lane&15  -> two 16B loads + shuffle.
// B (Kx16, packed [N][Kpad]): lanes 0-15 hold K 0..15, lanes 16-31 hold
// K 16..31 for col N = lane&15 -> one contiguous 32B load.
// ---------------------------------------------------------------------------
__device__ __forceinline__ v16h load_a_frag(const half_t* base, int stride,
                                            int row_tile, int kt, int lane) {
  int m = lane & 15;
  int kh = (lane < 16) ? 0 : 16;
  const half_t* p = base + (row_tile * 16 + m) * stride + kt * 32 + kh;
  v8h lo = *(const v8h*)p;
  v8h hi = *(const v8h*)(p + 8);
  return __builtin_shufflevector(lo, hi, 0,1,2,3,4,5,6,7,8,9,10,11,12,13,14,15);
}

__device__ __forceinline__ v16h load_b_frag(const half_t* W, int stride,
                                            int col_tile, int kt, int lane) {
  int n = lane & 15;
  int kh = (lane < 16) ? 0 : 16;
  const half_t* p = W + (col_tile * 16 + n) * stride + kt * 32 + kh;
  return *(const v16h*)p;
}

// ---------------------------------------------------------------------------
// Fused DeepFM kernel: 128 rows per WG, 8 waves
// ---------------------------------------------------------------------------
__global__ __launch_bounds__(256, 2)
void deepfm_fused(const float* __restrict__ nf,      // [N,13]
                  const int*   __restrict__ cidx,    // [N,16]
                  const float* __restrict__ emb1d,   // [16,V,1]
                  const float* __restrict__ emb2d,   // [16,V,16]
                  const float* __restrict__ dense_w, // [13]
                  const float* __restrict__ dense_b, // [1]
                  const float* __restrict__ b1,      // [256]
                  const float* __restrict__ b2,      // [128]
                  const float* __restrict__ bo,      // [128]
                  const half_t* __restrict__ W1p,    // [256][288]
                  const half_t* __restrict__ W2p,    // [128][256]
                  const half_t* __restrict__ W3p,    // [128][160]
                  float* __restrict__ out)           // [N,128]
{
  extern __shared__ char smem[];
  half_t* A1  = (half_t*)(smem + A1_OFF);   // 128 x 288
  half_t* A2  = (half_t*)(smem + A2_OFF);   // 128 x 256
  half_t* A3  = (half_t*)(smem + A1_OFF);   // 128 x 160 (aliases A1; A1 dead)
  float*  fm1 = (float*)(smem + FM_OFF);
  float*  fm2 = fm1 + ROWS;

  const int tid  = threadIdx.x;
  const int lane = tid & 31;
  const int wave = tid >> 5;
  const int r0   = blockIdx.x * ROWS;

  // ---- Phase 0a: gather emb2d rows (L2-resident) -> f16 in A1[:, 0:256] ----
  #pragma unroll
  for (int it = 0; it < 8; ++it) {
    int task = tid + it * 256;          // 0..2047 = 128 rows x 16 fields
    int rl = task >> 4;
    int f  = task & 15;
    int idx = cidx[(size_t)(r0 + rl) * 16 + f];
    const float* src = emb2d + ((size_t)f * FVOC + (size_t)idx) * 16;
    v8h h0, h1;
    #pragma unroll
    for (int d = 0; d < 8; ++d) h0[d] = (half_t)src[d];
    #pragma unroll
    for (int d = 0; d < 8; ++d) h1[d] = (half_t)src[8 + d];
    v8h* dst = (v8h*)(A1 + rl * K1 + f * 16);
    dst[0] = h0;
    dst[1] = h1;
  }
  __syncthreads();

  // ---- Phase 0b: dense tail cols + FM 1st/2nd order scalars ----
  if (tid < ROWS) {
    int rl = tid;
    const float* nrow = nf + (size_t)(r0 + rl) * 13;
    float dot = dense_b[0];
    #pragma unroll
    for (int j = 0; j < 13; ++j) {
      float v = nrow[j];
      A1[rl * K1 + 256 + j] = (half_t)v;
      dot += v * dense_w[j];
    }
    #pragma unroll
    for (int j = 269; j < K1; ++j) A1[rl * K1 + j] = (half_t)0.f;

    float s1 = 0.f;
    #pragma unroll
    for (int f = 0; f < 16; ++f) {
      int idx = cidx[(size_t)(r0 + rl) * 16 + f];
      s1 += emb1d[(size_t)f * FVOC + idx];
    }
    fm1[rl] = s1 + dot;

    // FM 2nd order: 0.5 * (sum(e)^2 - sum(e^2)) summed over D
    float sumd[16];
    #pragma unroll
    for (int d = 0; d < 16; ++d) sumd[d] = 0.f;
    float sq = 0.f;
    for (int f = 0; f < 16; ++f) {
      const half_t* e = A1 + rl * K1 + f * 16;
      #pragma unroll
      for (int d = 0; d < 16; ++d) {
        float v = (float)e[d];
        sumd[d] += v;
        sq += v * v;
      }
    }
    float s2 = 0.f;
    #pragma unroll
    for (int d = 0; d < 16; ++d) s2 += sumd[d] * sumd[d];
    fm2[rl] = 0.5f * (s2 - sq);
  }
  __syncthreads();

  // ---- GEMM1: A1[128x288] x W1p -> relu -> A2[128x256] ----
  for (int t = wave; t < (ROWS / 16) * (N1 / 16); t += 8) {
    int rt = t >> 4;
    int ct = t & 15;
    // prefetch next tile's B panel into cache
    if (t + 8 < (ROWS / 16) * (N1 / 16)) {
      int ctn = (t + 8) & 15;
      __builtin_prefetch(W1p + (ctn * 16 + (lane & 15)) * K1, 0, 1);
    }
    v8f acc = {};
    #pragma unroll
    for (int kt = 0; kt < K1 / 32; ++kt) {
      v16h a = load_a_frag(A1, K1, rt, kt, lane);
      v16h b = load_b_frag(W1p, K1, ct, kt, lane);
      acc = __builtin_amdgcn_wmma_f32_16x16x32_f16(
          false, a, false, b, (short)0, acc, false, false);
    }
    int n = ct * 16 + (lane & 15);
    float bias = b1[n];
    int mbase = rt * 16 + ((lane < 16) ? 0 : 8);
    #pragma unroll
    for (int i = 0; i < 8; ++i) {
      float v = acc[i] + bias;
      v = v > 0.f ? v : 0.f;
      A2[(mbase + i) * N1 + n] = (half_t)v;
    }
  }
  __syncthreads();   // A1 fully consumed; A3 (aliasing A1) now writable

  // ---- A3 scalar columns: fm1, fm2, zero pad 130..159 ----
  if (tid < ROWS) {
    A3[tid * K3 + 0] = (half_t)fm1[tid];
    A3[tid * K3 + 1] = (half_t)fm2[tid];
    #pragma unroll
    for (int k = 130; k < K3; ++k) A3[tid * K3 + k] = (half_t)0.f;
  }

  // ---- GEMM2: A2[128x256] x W2p -> relu -> A3 cols 2..129 ----
  for (int t = wave; t < (ROWS / 16) * (N2 / 16); t += 8) {
    int rt = t >> 3;
    int ct = t & 7;
    v8f acc = {};
    #pragma unroll
    for (int kt = 0; kt < K2 / 32; ++kt) {
      v16h a = load_a_frag(A2, K2, rt, kt, lane);
      v16h b = load_b_frag(W2p, K2, ct, kt, lane);
      acc = __builtin_amdgcn_wmma_f32_16x16x32_f16(
          false, a, false, b, (short)0, acc, false, false);
    }
    int n = ct * 16 + (lane & 15);
    float bias = b2[n];
    int mbase = rt * 16 + ((lane < 16) ? 0 : 8);
    #pragma unroll
    for (int i = 0; i < 8; ++i) {
      float v = acc[i] + bias;
      v = v > 0.f ? v : 0.f;
      A3[(mbase + i) * K3 + 2 + n] = (half_t)v;
    }
  }
  __syncthreads();

  // ---- GEMM3: A3[128x160] x W3p -> relu -> out (fp32) ----
  for (int t = wave; t < (ROWS / 16) * (N3 / 16); t += 8) {
    int rt = t >> 3;
    int ct = t & 7;
    v8f acc = {};
    #pragma unroll
    for (int kt = 0; kt < K3 / 32; ++kt) {
      v16h a = load_a_frag(A3, K3, rt, kt, lane);
      v16h b = load_b_frag(W3p, K3, ct, kt, lane);
      acc = __builtin_amdgcn_wmma_f32_16x16x32_f16(
          false, a, false, b, (short)0, acc, false, false);
    }
    int n = ct * 16 + (lane & 15);
    float bias = bo[n];
    int mbase = rt * 16 + ((lane < 16) ? 0 : 8);
    #pragma unroll
    for (int i = 0; i < 8; ++i) {
      float v = acc[i] + bias;
      v = v > 0.f ? v : 0.f;
      out[(size_t)(r0 + mbase + i) * N3 + n] = v;
    }
  }
}

// ---------------------------------------------------------------------------
extern "C" void kernel_launch(void* const* d_in, const int* in_sizes, int n_in,
                              void* d_out, int out_size, void* d_ws, size_t ws_size,
                              hipStream_t stream) {
  (void)in_sizes; (void)n_in; (void)out_size; (void)ws_size;
  const float* nf      = (const float*)d_in[0];
  const int*   cidx    = (const int*)  d_in[1];
  const float* emb1d   = (const float*)d_in[2];
  const float* emb2d   = (const float*)d_in[3];
  const float* dense_w = (const float*)d_in[4];
  const float* dense_b = (const float*)d_in[5];
  const float* w1      = (const float*)d_in[6];
  const float* b1      = (const float*)d_in[7];
  const float* w2      = (const float*)d_in[8];
  const float* b2      = (const float*)d_in[9];
  const float* wo      = (const float*)d_in[10];
  const float* bo      = (const float*)d_in[11];

  half_t* W1p = (half_t*)((char*)d_ws + W1P_OFF);
  half_t* W2p = (half_t*)((char*)d_ws + W2P_OFF);
  half_t* W3p = (half_t*)((char*)d_ws + W3P_OFF);

  // 1) pack/pad weights to f16 [N][Kpad]
  pack_weights<<<dim3((N1 * K1 + 255) / 256), dim3(256), 0, stream>>>(
      w1, w2, wo, W1p, W2p, W3p);

  // 2) fused DeepFM
  deepfm_fused<<<dim3(NROWS / ROWS), dim3(256), LDS_BYTES, stream>>>(
      nf, cidx, emb1d, emb2d, dense_w, dense_b, b1, b2, bo,
      W1p, W2p, W3p, (float*)d_out);
}
